// VectorQuantizer_8796093022597
// MI455X (gfx1250) — matrix operand order repaired and checked by reading, not compiled, not run on hardware
//
#include <hip/hip_runtime.h>

typedef __attribute__((ext_vector_type(16))) __bf16 v16bf;
typedef __attribute__((ext_vector_type(8)))  float  v8f;

#define N_ROWS      (32 * 4096)     // B*T
#define D_DIM       64
#define K_CODES     512
#define ROWS_PER_WG 256             // 8 waves x 2 M-tiles x 16 rows
#define IDX_OFF     ((long)N_ROWS * D_DIM)
#define LOSS_OFF    (IDX_OFF + N_ROWS)

__global__ void vq_zero(float* ws) { ws[0] = 0.0f; }

__global__ void vq_finalize(const float* __restrict__ ws, float* __restrict__ out) {
  out[LOSS_OFF] = 0.25f * ws[0] / 8388608.0f;   // / (N_ROWS * D_DIM)
}

__global__ __launch_bounds__(256)
void vq_main(const float* __restrict__ z, const float* __restrict__ Wg,
             float* __restrict__ out, float* __restrict__ ws) {
  __shared__ __bf16 Wlds[K_CODES * D_DIM];   // 64 KB bf16 codebook
  __shared__ float  wnorm[K_CODES];          // 2 KB ||w||^2 (fp32)

  const int t = threadIdx.x;

  // ---- Stage codebook: fp32 -> bf16 into LDS, compute ||w||^2 ----
  for (int r = t; r < K_CODES; r += 256) {
    const float4* wr4 = (const float4*)(Wg + r * D_DIM);
    float s = 0.0f;
#pragma unroll
    for (int q = 0; q < 16; ++q) {
      float4 v = wr4[q];
      s += v.x * v.x + v.y * v.y + v.z * v.z + v.w * v.w;
      int b = r * D_DIM + q * 4;
      Wlds[b + 0] = (__bf16)v.x;
      Wlds[b + 1] = (__bf16)v.y;
      Wlds[b + 2] = (__bf16)v.z;
      Wlds[b + 3] = (__bf16)v.w;
    }
    wnorm[r] = s;
  }
  __syncthreads();

  const int lane = t & 31;
  const int wave = t >> 5;       // 8 waves, each owns 2 M-tiles (32 rows)
  const int m16  = lane & 15;
  const int half = lane >> 4;

  // ---- Load A fragments for both M-tiles (16x32 bf16 x2 chunks each) ----
  // half=0 lanes hold K 0..15; half=1 lanes hold K 16..31
  const long rowBase = (long)blockIdx.x * ROWS_PER_WG + wave * 32;
  const float* zr0 = z + (rowBase + m16)      * D_DIM;   // tile 0: rows 0..15
  const float* zr1 = z + (rowBase + 16 + m16) * D_DIM;   // tile 1: rows 16..31
  v16bf a0L, a1L, a0H, a1H;
#pragma unroll
  for (int i = 0; i < 8; ++i) {
    a0L[i]     = (__bf16)zr0[half * 16 + i];
    a0L[i + 8] = (__bf16)zr0[half * 16 + 8 + i];
    a1L[i]     = (__bf16)zr0[32 + half * 16 + i];
    a1L[i + 8] = (__bf16)zr0[32 + half * 16 + 8 + i];
    a0H[i]     = (__bf16)zr1[half * 16 + i];
    a0H[i + 8] = (__bf16)zr1[half * 16 + 8 + i];
    a1H[i]     = (__bf16)zr1[32 + half * 16 + i];
    a1H[i + 8] = (__bf16)zr1[32 + half * 16 + 8 + i];
  }

  // Packed argmin keys: key = (bits(dist+4) & ~0x1FF) | code, reduced by min_u32.
  // +4 bias keeps dist positive so float order == unsigned bit order; the bias
  // is per-row constant so it cannot change the argmin. Low 9 bits carry the
  // code, so ties resolve to the lowest index (matches jnp.argmin).
  unsigned key0[8], key1[8];
#pragma unroll
  for (int r = 0; r < 8; ++r) { key0[r] = 0xFFFFFFFFu; key1[r] = 0xFFFFFFFFu; }

  const __bf16* wbase = &Wlds[m16 * D_DIM + half * 16];

  for (int nt = 0; nt < K_CODES / 16; ++nt) {
    const int code = nt * 16 + m16;               // this lane's column/code
    const __bf16* wrow = wbase + nt * 16 * D_DIM;
    v16bf b0 = *(const v16bf*)(wrow);             // K = half*16 .. +15
    v16bf b1 = *(const v16bf*)(wrow + 32);        // K = 32+half*16 .. +15
    float wn4 = wnorm[code] + 4.0f;

    v8f c0 = {}, c1 = {};
    c0 = __builtin_amdgcn_wmma_f32_16x16x32_bf16(false, a0L, false, b0,
                                                 (short)0, c0, false, false);
    c1 = __builtin_amdgcn_wmma_f32_16x16x32_bf16(false, a0H, false, b0,
                                                 (short)0, c1, false, false);
    c0 = __builtin_amdgcn_wmma_f32_16x16x32_bf16(false, a1L, false, b1,
                                                 (short)0, c0, false, false);
    c1 = __builtin_amdgcn_wmma_f32_16x16x32_bf16(false, a1H, false, b1,
                                                 (short)0, c1, false, false);

#pragma unroll
    for (int r = 0; r < 8; ++r) {
      float d0 = __builtin_fmaf(-2.0f, c0[r], wn4);
      float d1 = __builtin_fmaf(-2.0f, c1[r], wn4);
      unsigned k0 = (__float_as_uint(d0) & 0xFFFFFE00u) | (unsigned)code;
      unsigned k1 = (__float_as_uint(d1) & 0xFFFFFE00u) | (unsigned)code;
      key0[r] = (k0 < key0[r]) ? k0 : key0[r];   // v_min_u32
      key1[r] = (k1 < key1[r]) ? k1 : key1[r];
    }
  }

  // ---- Argmin reduce across the 16 lanes holding each row ----
#pragma unroll
  for (int off = 8; off >= 1; off >>= 1) {
#pragma unroll
    for (int r = 0; r < 8; ++r) {
      unsigned o0 = __shfl_xor(key0[r], off, 32);
      unsigned o1 = __shfl_xor(key1[r], off, 32);
      key0[r] = (o0 < key0[r]) ? o0 : key0[r];
      key1[r] = (o1 < key1[r]) ? o1 : key1[r];
    }
  }
  // tile0 rows 0..7 in lanes 0..15, rows 8..15 in lanes 16..31; same for tile1

  // ---- Gather z_q (exact fp32 from global W), write indices, exact loss ----
  float sq = 0.0f;
#pragma unroll
  for (int rr = 0; rr < 32; ++rr) {
    const int slot = rr & 7;
    const int src  = ((rr >> 3) & 1) * 16;
    const unsigned key = (rr < 16) ? __shfl(key0[slot], src, 32)
                                   : __shfl(key1[slot], src, 32);
    const int idx = (int)(key & 511u);
    const long g  = rowBase + rr;
    const float2* zp = (const float2*)(z  + g * D_DIM);
    const float2* wp = (const float2*)(Wg + (long)idx * D_DIM);
    float2 zv = zp[lane];
    float2 wv = wp[lane];
    ((float2*)(out + g * D_DIM))[lane] = wv;
    float d0 = zv.x - wv.x, d1 = zv.y - wv.y;
    sq += d0 * d0 + d1 * d1;
    if (lane == 0) out[IDX_OFF + g] = (float)idx;
  }
#pragma unroll
  for (int off = 16; off >= 1; off >>= 1) sq += __shfl_xor(sq, off, 32);
  if (lane == 0) atomicAdd(ws, sq);
}

extern "C" void kernel_launch(void* const* d_in, const int* in_sizes, int n_in,
                              void* d_out, int out_size, void* d_ws, size_t ws_size,
                              hipStream_t stream) {
  const float* z  = (const float*)d_in[0];   // [32,4096,64] fp32
  const float* Wc = (const float*)d_in[1];   // [512,64] fp32
  float* out = (float*)d_out;                // z_q (N*D) | indices (N) | loss (1)
  float* ws  = (float*)d_ws;

  vq_zero<<<1, 1, 0, stream>>>(ws);
  vq_main<<<N_ROWS / ROWS_PER_WG, 256, 0, stream>>>(z, Wc, out, ws);
  vq_finalize<<<1, 1, 0, stream>>>(ws, out);
}
